// AttentionLayer_2637109920089
// MI455X (gfx1250) — matrix operand order repaired and checked by reading, not compiled, not run on hardware
//
#include <hip/hip_runtime.h>

typedef __attribute__((ext_vector_type(16))) __bf16 v16bf;
typedef __attribute__((ext_vector_type(8)))  float  v8f;

constexpr int BB = 32;    // batch
constexpr int TT = 4096;  // time
constexpr int HH = 256;   // hidden
constexpr int UU = 128;   // units

// ---------------------------------------------------------------------------
// Kernel 1: bias[b,u] = dec[b,:] @ Uw[:,u] + Ub[u] + Wb[u]   (B*U = 4096 vals)
// ---------------------------------------------------------------------------
__global__ void k_bias(const float* __restrict__ dec, const float* __restrict__ Uw,
                       const float* __restrict__ Ub,  const float* __restrict__ Wb,
                       float* __restrict__ bias) {
    const int idx = blockIdx.x * 256 + threadIdx.x;   // [0, 4096)
    const int b = idx >> 7, u = idx & 127;
    float acc = Ub[u] + Wb[u];
#pragma unroll 8
    for (int h = 0; h < HH; ++h) acc += dec[b * HH + h] * Uw[h * UU + u];
    bias[idx] = acc;
}

// ---------------------------------------------------------------------------
// Kernel 2: pack Ww (fp32 [256,128] row-major) into bf16 WMMA B-operand tiles.
// Layout per (kt,nt) tile: lane n (0..15) = column N=nt*16+n holding K=kt*32+{0..7,16..23}
// (2 bf16/VGPR), lane n+16 same column holding K=kt*32+{8..15,24..31}.
// Linear index: ((kt*8+nt)*32 + lane)*16 + j  with K = (j>>3)*16 + (lane>>4)*8 + (j&7).
// ---------------------------------------------------------------------------
__global__ void k_pack(const float* __restrict__ Ww, unsigned short* __restrict__ wp) {
    const int idx  = blockIdx.x * 256 + threadIdx.x;  // [0, 32768)
    const int j    = idx & 15;
    const int lane = (idx >> 4) & 31;
    const int nt   = (idx >> 9) & 7;
    const int kt   = idx >> 12;
    const int k = kt * 32 + ((lane >> 4) << 3) + ((j >> 3) << 4) + (j & 7);
    const int u = nt * 16 + (lane & 15);
    union { __bf16 h; unsigned short s; } cv;
    cv.h = (__bf16)Ww[k * UU + u];
    wp[idx] = cv.s;
}

// ---------------------------------------------------------------------------
// Kernel 3: WMMA score kernel.
// Block = 256 threads = 8 waves; wave w handles rows t0 = bx*128 + w*16 .. +15.
// scores[b,t] = sum_u tanh( enc[b,t,:]·Ww[:,u] + bias[b,u] ) * vw[u]  (+vb)
// ---------------------------------------------------------------------------
__global__ void __launch_bounds__(256) k_score(
        const float* __restrict__ enc, const unsigned short* __restrict__ wp,
        const float* __restrict__ bias, const float* __restrict__ vw,
        const float* __restrict__ vb, float* __restrict__ scores) {
    __shared__ float4 ldsW[4096];   // 64 KB: packed bf16 Ww, copied verbatim
    {
        const float4* g = reinterpret_cast<const float4*>(wp);
        for (int i = threadIdx.x; i < 4096; i += 256) ldsW[i] = g[i];
    }
    __syncthreads();

    const int b     = blockIdx.y;
    const int lane  = threadIdx.x & 31;
    const int wave  = threadIdx.x >> 5;
    const int t0    = blockIdx.x * 128 + wave * 16;
    const int mrow  = lane & 15;      // A-row / C-column index within tile
    const int khalf = lane >> 4;      // 0 or 1

    // ---- Load A: 16 rows x 256 cols fp32 -> bf16, 8 K-tiles of 32.
    // A layout (16-bit, 16x32): lanes 0-15 hold K={0..7,16..23}, lanes 16-31
    // hold K={8..15,24..31}; element j -> K = (j>>3)*16 + khalf*8 + (j&7).
    const size_t rowBase = ((size_t)b * TT + t0 + mrow) * HH;
    v16bf aT[8];
#pragma unroll
    for (int kt = 0; kt < 8; ++kt) {
        const int h0 = kt * 32 + khalf * 8;
        const float4 f0 = *(const float4*)(enc + rowBase + h0);
        const float4 f1 = *(const float4*)(enc + rowBase + h0 + 4);
        const float4 f2 = *(const float4*)(enc + rowBase + h0 + 16);
        const float4 f3 = *(const float4*)(enc + rowBase + h0 + 20);
        union { v16bf v; __bf16 e[16]; } ua;
        ua.e[0]  = (__bf16)f0.x; ua.e[1]  = (__bf16)f0.y;
        ua.e[2]  = (__bf16)f0.z; ua.e[3]  = (__bf16)f0.w;
        ua.e[4]  = (__bf16)f1.x; ua.e[5]  = (__bf16)f1.y;
        ua.e[6]  = (__bf16)f1.z; ua.e[7]  = (__bf16)f1.w;
        ua.e[8]  = (__bf16)f2.x; ua.e[9]  = (__bf16)f2.y;
        ua.e[10] = (__bf16)f2.z; ua.e[11] = (__bf16)f2.w;
        ua.e[12] = (__bf16)f3.x; ua.e[13] = (__bf16)f3.y;
        ua.e[14] = (__bf16)f3.z; ua.e[15] = (__bf16)f3.w;
        aT[kt] = ua.v;
    }

    float partial[8] = {0.f, 0.f, 0.f, 0.f, 0.f, 0.f, 0.f, 0.f};

#pragma unroll
    for (int nt = 0; nt < 8; ++nt) {
        v8f acc = {};
#pragma unroll
        for (int kt = 0; kt < 8; ++kt) {
            const v16bf bm = *reinterpret_cast<const v16bf*>(
                &ldsW[(((kt << 3) + nt) * 32 + lane) * 2]);
            acc = __builtin_amdgcn_wmma_f32_16x16x32_bf16(
                false, aT[kt], false, bm, (short)0, acc, false, false);
        }
        // C layout: VGPR r, lanes 0-15 -> (M=r, N=lane); lanes 16-31 -> (M=r+8).
        const float bv = bias[b * UU + nt * 16 + mrow];
        const float vv = vw[nt * 16 + mrow];
        union { v8f v; float e[8]; } uc; uc.v = acc;
#pragma unroll
        for (int r = 0; r < 8; ++r) partial[r] += tanhf(uc.e[r] + bv) * vv;
    }

    // Reduce over N (16 lanes within each half-wave; xor offsets keep halves apart)
#pragma unroll
    for (int r = 0; r < 8; ++r) {
        float p = partial[r];
        p += __shfl_xor(p, 1);
        p += __shfl_xor(p, 2);
        p += __shfl_xor(p, 4);
        p += __shfl_xor(p, 8);
        partial[r] = p;
    }
    const float vbv = vb[0];
    if (mrow == 0) {          // lane 0 -> rows t0..t0+7, lane 16 -> t0+8..t0+15
#pragma unroll
        for (int r = 0; r < 8; ++r)
            scores[(size_t)b * TT + t0 + khalf * 8 + r] = partial[r] + vbv;
    }
}

// ---------------------------------------------------------------------------
// Kernel 4: softmax over T per batch row; writes weights into d_out.
// ---------------------------------------------------------------------------
__global__ void k_softmax(const float* __restrict__ scores, float* __restrict__ wout) {
    __shared__ float red[256];
    const int b = blockIdx.x, tid = threadIdx.x;
    float v[16];
    float m = -3.402823466e38f;
#pragma unroll
    for (int i = 0; i < 16; ++i) {
        v[i] = scores[(size_t)b * TT + i * 256 + tid];
        m = fmaxf(m, v[i]);
    }
    red[tid] = m; __syncthreads();
    for (int s = 128; s > 0; s >>= 1) {
        if (tid < s) red[tid] = fmaxf(red[tid], red[tid + s]);
        __syncthreads();
    }
    m = red[0]; __syncthreads();
    float sum = 0.f;
#pragma unroll
    for (int i = 0; i < 16; ++i) { v[i] = __expf(v[i] - m); sum += v[i]; }
    red[tid] = sum; __syncthreads();
    for (int s = 128; s > 0; s >>= 1) {
        if (tid < s) red[tid] += red[tid + s];
        __syncthreads();
    }
    const float inv = 1.f / red[0];
#pragma unroll
    for (int i = 0; i < 16; ++i)
        wout[(size_t)b * TT + i * 256 + tid] = v[i] * inv;
}

// ---------------------------------------------------------------------------
// Kernel 5: partial context: block (chunk, b) sums 256 T-rows; thread = h col.
// Fully coalesced second pass over encoder_outputs; no atomics.
// ---------------------------------------------------------------------------
__global__ void k_ctx_partial(const float* __restrict__ enc, const float* __restrict__ w,
                              float* __restrict__ part) {
    __shared__ float wsh[256];
    const int b = blockIdx.y, chunk = blockIdx.x, h = threadIdx.x;
    const int tbase = chunk * 256;
    wsh[h] = w[(size_t)b * TT + tbase + h];
    __syncthreads();
    float acc = 0.f;
    const float* ep = enc + ((size_t)b * TT + tbase) * HH + h;
#pragma unroll 4
    for (int t = 0; t < 256; ++t) acc += wsh[t] * ep[(size_t)t * HH];
    part[((size_t)b * 16 + chunk) * HH + h] = acc;
}

// ---------------------------------------------------------------------------
// Kernel 6: reduce 16 partials -> context[b,h] in d_out.
// ---------------------------------------------------------------------------
__global__ void k_ctx_reduce(const float* __restrict__ part, float* __restrict__ ctx) {
    const int idx = blockIdx.x * 256 + threadIdx.x;   // [0, 8192)
    const int b = idx >> 8, h = idx & 255;
    float acc = 0.f;
#pragma unroll
    for (int c = 0; c < 16; ++c) acc += part[((size_t)b * 16 + c) * HH + h];
    ctx[idx] = acc;
}

// ---------------------------------------------------------------------------
extern "C" void kernel_launch(void* const* d_in, const int* in_sizes, int n_in,
                              void* d_out, int out_size, void* d_ws, size_t ws_size,
                              hipStream_t stream) {
    const float* dec = (const float*)d_in[0];   // [32,256]
    const float* enc = (const float*)d_in[1];   // [32,4096,256]
    const float* Ww  = (const float*)d_in[2];   // [256,128]
    const float* Wb  = (const float*)d_in[3];   // [128]
    const float* Uw  = (const float*)d_in[4];   // [256,128]
    const float* Ub  = (const float*)d_in[5];   // [128]
    const float* vw  = (const float*)d_in[6];   // [128]
    const float* vb  = (const float*)d_in[7];   // [1]

    float* out  = (float*)d_out;
    float* ctx  = out;                 // [32,256]
    float* attw = out + BB * HH;       // [32,4096]

    char* ws = (char*)d_ws;
    float*          bias   = (float*)(ws);                               // 16 KB
    float*          scores = (float*)(ws + 16384);                       // 512 KB
    float*          part   = (float*)(ws + 16384 + 524288);              // 512 KB
    unsigned short* wp     = (unsigned short*)(ws + 16384 + 2 * 524288); // 64 KB

    k_bias       <<<16,            256, 0, stream>>>(dec, Uw, Ub, Wb, bias);
    k_pack       <<<128,           256, 0, stream>>>(Ww, wp);
    k_score      <<<dim3(32, 32),  256, 0, stream>>>(enc, wp, bias, vw, vb, scores);
    k_softmax    <<<32,            256, 0, stream>>>(scores, attw);
    k_ctx_partial<<<dim3(16, 32),  256, 0, stream>>>(enc, attw, part);
    k_ctx_reduce <<<32,            256, 0, stream>>>(part, ctx);
}
